// QuantizedConv2d_4750233829797
// MI455X (gfx1250) — 8-bit matrix operand order repaired and checked by reading, compiled, not run on hardware
//
#include <hip/hip_runtime.h>

typedef __attribute__((ext_vector_type(8))) int v8i;

#define N_IMG 32
#define CIN   128
#define HDIM  56
#define WDIM  56
#define COUT  256
#define KTOT  1152
#define SP    3136
#define TS    (N_IMG * SP)
#define PADB  0xFDFDFDFDu

__global__ void pack_x_kernel(const int* __restrict__ x,
                              unsigned int* __restrict__ xp) {
  int d = blockIdx.x * 256 + threadIdx.x;
  const int total = N_IMG * (CIN / 4) * SP;
  if (d >= total) return;
  int n   = d / ((CIN / 4) * SP);
  int rem = d - n * ((CIN / 4) * SP);
  int cq  = rem / SP;
  int hw  = rem - cq * SP;
  int base = (n * CIN + cq * 4) * SP + hw;
  unsigned int b0 = (unsigned int)x[base        ] & 255u;
  unsigned int b1 = (unsigned int)x[base +   SP ] & 255u;
  unsigned int b2 = (unsigned int)x[base + 2*SP ] & 255u;
  unsigned int b3 = (unsigned int)x[base + 3*SP ] & 255u;
  xp[d] = b0 | (b1 << 8) | (b2 << 16) | (b3 << 24);
}

__global__ void pack_w_kernel(const int* __restrict__ w,
                              unsigned int* __restrict__ wp) {
  int d = blockIdx.x * 256 + threadIdx.x;
  const int total = COUT * (KTOT / 4);
  if (d >= total) return;
  int cout = d / (KTOT / 4);
  int rem  = d - cout * (KTOT / 4);
  int k0   = rem * 4;
  int rs   = k0 >> 7;
  int c    = k0 & 127;
  unsigned int b0 = (unsigned int)w[(cout * CIN + c + 0) * 9 + rs] & 255u;
  unsigned int b1 = (unsigned int)w[(cout * CIN + c + 1) * 9 + rs] & 255u;
  unsigned int b2 = (unsigned int)w[(cout * CIN + c + 2) * 9 + rs] & 255u;
  unsigned int b3 = (unsigned int)w[(cout * CIN + c + 3) * 9 + rs] & 255u;
  wp[d] = b0 | (b1 << 8) | (b2 << 16) | (b3 << 24);
}

__launch_bounds__(256, 2)
__global__ void qconv_wmma_kernel(const unsigned char* __restrict__ xp,
                                  const unsigned char* __restrict__ wp,
                                  const int* __restrict__ bias,
                                  const float* __restrict__ wscale,
                                  float* __restrict__ out) {
  __shared__ __align__(16) unsigned char lA[64 * 80];
  __shared__ __align__(16) unsigned char lB[128 * 80];
  __shared__ int   sBias[64];
  __shared__ float sWs[64];

  const int tid  = threadIdx.x;
  const int lane = tid & 31;
  const int wave = tid >> 5;
  const int mt   = wave >> 2;
  const int nt   = wave & 3;

  const int p0    = blockIdx.x * 128;
  const int cout0 = blockIdx.y * 64;

  if (tid < 64) {
    sBias[tid] = bias[cout0 + tid];
    sWs[tid]   = wscale[cout0 + tid];
  }

  const int arow = tid >> 2;
  const int achk = tid & 3;
  const int bpos0 = tid >> 1;
  const int t0pos = (tid)        >> 2, t0chk = (tid)        & 3;
  const int t1pos = (tid + 256)  >> 2, t1chk = (tid + 256)  & 3;
  (void)bpos0;

  int fimg0, foh0, fow0, fimg1, foh1, fow1;
  {
    int fp = p0 + t0pos;
    fimg0 = fp / SP; int s = fp - fimg0 * SP; foh0 = s / WDIM; fow0 = s - foh0 * WDIM;
    fp = p0 + t1pos;
    fimg1 = fp / SP; s = fp - fimg1 * SP; foh1 = s / WDIM; fow1 = s - foh1 * WDIM;
  }

  auto loadA = [&](int rs, int ch) -> int4 {
    return *(const int4*)(wp + (size_t)(cout0 + arow) * KTOT +
                          rs * 128 + ch * 64 + achk * 16);
  };
  auto loadB = [&](int kh, int kw, int ch, int img, int oh, int ow,
                   int chk) -> int4 {
    int h = oh + kh - 1;
    int w = ow + kw - 1;
    int4 v;
    if ((unsigned)h < (unsigned)HDIM && (unsigned)w < (unsigned)WDIM) {
      const unsigned int* src = (const unsigned int*)xp +
          ((size_t)(img * (CIN / 4) + ch * 16 + chk * 4) * SP + h * WDIM + w);
      v.x = (int)src[0];
      v.y = (int)src[SP];
      v.z = (int)src[2 * SP];
      v.w = (int)src[3 * SP];
    } else {
      v.x = v.y = v.z = v.w = (int)PADB;
    }
    return v;
  };

  v8i acc[2][2];
#pragma unroll
  for (int i = 0; i < 2; ++i)
#pragma unroll
    for (int j = 0; j < 2; ++j)
      acc[i][j] = (v8i){0, 0, 0, 0, 0, 0, 0, 0};

  const unsigned char* aBase0 =
      &lA[(mt * 32 + 0 * 16 + (lane & 15)) * 80 + (lane >> 4) * 8];
  const unsigned char* aBase1 = aBase0 + 16 * 80;
  const unsigned char* bBase0 =
      &lB[(nt * 32 + 0 * 16 + (lane & 15)) * 80 + (lane >> 4) * 8];
  const unsigned char* bBase1 = bBase0 + 16 * 80;

  int4 apref = loadA(0, 0);
  int4 bpref0 = loadB(0, 0, 0, fimg0, foh0, fow0, t0chk);
  int4 bpref1 = loadB(0, 0, 0, fimg1, foh1, fow1, t1chk);

  for (int ks = 0; ks < 18; ++ks) {
    __syncthreads();

    *(int4*)(&lA[arow  * 80 + achk  * 16]) = apref;
    *(int4*)(&lB[t0pos * 80 + t0chk * 16]) = bpref0;
    *(int4*)(&lB[t1pos * 80 + t1chk * 16]) = bpref1;

    if (ks < 17) {
      const int nrs = (ks + 1) >> 1;
      const int nch = (ks + 1) & 1;
      const int nkh = nrs / 3;
      const int nkw = nrs - nkh * 3;
      apref  = loadA(nrs, nch);
      bpref0 = loadB(nkh, nkw, nch, fimg0, foh0, fow0, t0chk);
      bpref1 = loadB(nkh, nkw, nch, fimg1, foh1, fow1, t1chk);
    }

    __syncthreads();

    v8i a0, a1;
#pragma unroll
    for (int v = 0; v < 8; ++v) {
      const int koff = (v & 1) * 4 + ((v >> 1) & 1) * 16 + (v >> 2) * 32;
      a0[v] = *(const int*)(aBase0 + koff);
      a1[v] = *(const int*)(aBase1 + koff);
    }
    v8i b0, b1;
#pragma unroll
    for (int v = 0; v < 8; ++v) {
      const int koff = (v & 1) * 4 + ((v >> 1) & 1) * 16 + (v >> 2) * 32;
      b0[v] = *(const int*)(bBase0 + koff);
      b1[v] = *(const int*)(bBase1 + koff);
    }

    acc[0][0] = __builtin_amdgcn_wmma_i32_16x16x64_iu8(true, a0, true, b0,
                                                       acc[0][0], false, false);
    acc[0][1] = __builtin_amdgcn_wmma_i32_16x16x64_iu8(true, a0, true, b1,
                                                       acc[0][1], false, false);
    acc[1][0] = __builtin_amdgcn_wmma_i32_16x16x64_iu8(true, a1, true, b0,
                                                       acc[1][0], false, false);
    acc[1][1] = __builtin_amdgcn_wmma_i32_16x16x64_iu8(true, a1, true, b1,
                                                       acc[1][1], false, false);
  }

  const int halfm = 8 * (lane >> 4);
#pragma unroll
  for (int j = 0; j < 2; ++j) {
    const int p   = p0 + nt * 32 + j * 16 + (lane & 15);
    const int img = p / SP;
    const int sp  = p - img * SP;
#pragma unroll
    for (int i = 0; i < 2; ++i) {
      const int lc0 = mt * 32 + i * 16 + halfm;
#pragma unroll
      for (int r = 0; r < 8; ++r) {
        const int lc = lc0 + r;
        const int cg = cout0 + lc;
        const float sc  = 0.5f * sWs[lc];
        const float val = (float)(acc[i][j][r] + 2 * sBias[lc]) * sc + 5.0f;
        float q = __builtin_rintf(val);
        q = fminf(fmaxf(q, -128.0f), 127.0f);
        out[(size_t)(img * COUT + cg) * SP + sp] = q;
      }
    }
  }
}

extern "C" void kernel_launch(void* const* d_in, const int* in_sizes, int n_in,
                              void* d_out, int out_size, void* d_ws,
                              size_t ws_size, hipStream_t stream) {
  const int*   x      = (const int*)d_in[0];
  const int*   weight = (const int*)d_in[1];
  const int*   bias   = (const int*)d_in[2];
  const float* wscale = (const float*)d_in[3];
  float*       out    = (float*)d_out;

  unsigned char* xp = (unsigned char*)d_ws;
  unsigned char* wp = xp + (size_t)N_IMG * CIN * SP;

  const int xd = N_IMG * (CIN / 4) * SP;
  pack_x_kernel<<<(xd + 255) / 256, 256, 0, stream>>>(x, (unsigned int*)xp);

  const int wd = COUT * (KTOT / 4);
  pack_w_kernel<<<(wd + 255) / 256, 256, 0, stream>>>(weight,
                                                      (unsigned int*)wp);

  dim3 grid(TS / 128, COUT / 64);
  qconv_wmma_kernel<<<grid, 256, 0, stream>>>(xp, wp, bias, wscale, out);
}
